// HCN_50964081935359
// MI455X (gfx1250) — matrix operand order repaired and checked by reading, not compiled, not run on hardware
//
#include <hip/hip_runtime.h>

#define EPSF   1e-15f
#define MAXN   0.99999f
#define NFLOOR 3.1622776601e-8f   // sqrt(1e-15)

typedef __attribute__((ext_vector_type(8)))  __bf16 v8bf;
typedef __attribute__((ext_vector_type(16))) __bf16 v16bf;
typedef __attribute__((ext_vector_type(8)))  float  v8f;

// ---------------- wave (32-lane) helpers ----------------
__device__ __forceinline__ float wsum(float v){
  v += __shfl_xor(v, 16, 32);
  v += __shfl_xor(v,  8, 32);
  v += __shfl_xor(v,  4, 32);
  v += __shfl_xor(v,  2, 32);
  v += __shfl_xor(v,  1, 32);
  return v;
}

template<int RP>
__device__ __forceinline__ float rdot(const float* a, const float* b){
  float s = 0.f;
#pragma unroll
  for (int i = 0; i < RP; ++i) s += a[i]*b[i];
  return wsum(s);
}
template<int RP>
__device__ __forceinline__ float rnorm(const float* a){
  return sqrtf(fmaxf(rdot<RP>(a, a), EPSF));
}
template<int RP>
__device__ __forceinline__ float projr(float* v, float n){
  if (n > MAXN){
    float s = MAXN / n;
#pragma unroll
    for (int i = 0; i < RP; ++i) v[i] *= s;
    return MAXN;
  }
  return n;
}
__device__ __forceinline__ float artanh_(float x){
  x = fminf(fmaxf(x, -1.f + 1e-5f), 1.f - 1e-5f);
  return 0.5f * logf((1.f + x) / (1.f - x));
}
// mobius_add: o = ((1+2xy+y2)x + (1-x2)y) / max(1+2xy+x2*y2, EPS)   (C=1)
template<int RP>
__device__ __forceinline__ void mob_add(const float* x, const float* y,
                                        float x2, float y2, float xy, float* o){
  float a   = 1.f + 2.f*xy + y2;
  float b   = 1.f - x2;
  float inv = 1.f / fmaxf(1.f + 2.f*xy + x2*y2, EPSF);
#pragma unroll
  for (int i = 0; i < RP; ++i) o[i] = (a*x[i] + b*y[i]) * inv;
}
// mobius_matvec epilogue: mx *= tanh(mxn/xn*artanh(xn))/mxn ; proj
template<int RP>
__device__ __forceinline__ float mv_scale(float* mx, float mxn, float xn){
  float t = tanhf(mxn / fmaxf(xn, NFLOOR) * artanh_(xn));
  float s = t / mxn;
#pragma unroll
  for (int i = 0; i < RP; ++i) mx[i] *= s;
  return projr<RP>(mx, fabsf(t));
}
// mobius_pointwise_mul epilogue on precomputed wx (||wx||=wxn, ||x||=xn)
template<int RP>
__device__ __forceinline__ float pw_scale(float* wx, float wxn, float xn){
  float t = tanhf(wxn / fmaxf(xn, NFLOOR) * artanh_(xn));
  float s = t / wxn;
#pragma unroll
  for (int i = 0; i < RP; ++i) wx[i] *= s;
  return projr<RP>(wx, fabsf(t));
}
__device__ __forceinline__ unsigned short f2bf(float f){
  unsigned u = __float_as_uint(f);
  unsigned r = (u + 0x7fffu + ((u >> 16) & 1u)) >> 16;
  return (unsigned short)r;
}
__device__ __forceinline__ float sigm(float x){ return 1.f / (1.f + __expf(-x)); }

// ---------------- WMMA fragment helpers (bf16, 16x16x32) ----------------
// A (16xK): lane = M row (lane&15); hi half: K = kb + hi*8 .. +7 and kb+16+hi*8 .. +7
__device__ __forceinline__ v16bf load_a_frag(const unsigned short* rowp, int kb, int hi){
  v8bf c0 = *reinterpret_cast<const v8bf*>(rowp + kb + hi*8);
  v8bf c1 = *reinterpret_cast<const v8bf*>(rowp + kb + 16 + hi*8);
  return __builtin_shufflevector(c0, c1, 0,1,2,3,4,5,6,7,8,9,10,11,12,13,14,15);
}
// B (Kx16): lane = N column (lane&15); hi half: K = kb + hi*8 .. +7 and kb+16+hi*8 .. +7
__device__ __forceinline__ v16bf load_b_frag(const unsigned short* rowp, int kb, int hi){
  v8bf c0 = *reinterpret_cast<const v8bf*>(rowp + kb + hi*8);
  v8bf c1 = *reinterpret_cast<const v8bf*>(rowp + kb + 16 + hi*8);
  return __builtin_shufflevector(c0, c1, 0,1,2,3,4,5,6,7,8,9,10,11,12,13,14,15);
}
__device__ __forceinline__ v8f wmma_bf16(v16bf a, v16bf b, v8f c){
  return __builtin_amdgcn_wmma_f32_16x16x32_bf16(false, a, false, b, (short)0, c, false, false);
}

// ---------------- kernel: f32 -> bf16 convert ----------------
__global__ void cvt_bf16_k(const float* __restrict__ src, unsigned short* __restrict__ dst, int n){
  int i = blockIdx.x * blockDim.x + threadIdx.x;
  if (i < n) dst[i] = f2bf(src[i]);
}

// ---------------- kernel: per-node prep (x_ball, bf16 copies, row norms) ----------------
__global__ void prep_nodes_k(const float* __restrict__ x, const float* __restrict__ h1,
                             const float* __restrict__ c1,
                             unsigned short* __restrict__ xbbf, unsigned short* __restrict__ h1bf,
                             unsigned short* __restrict__ c1bf,
                             float* __restrict__ xbn, float* __restrict__ h1n,
                             float* __restrict__ c1n, int N){
  int wid  = blockIdx.x * 8 + (threadIdx.x >> 5);
  int lane = threadIdx.x & 31;
  if (wid >= N) return;
  float v[8];
  // x_ball = proj(tanh(|x|) * x / |x|)
#pragma unroll
  for (int i = 0; i < 8; ++i) v[i] = x[(size_t)wid*256 + i*32 + lane];
  float un = rnorm<8>(v);
  float t  = tanhf(un);
  float s  = t / un;
#pragma unroll
  for (int i = 0; i < 8; ++i) v[i] *= s;
  float nb = projr<8>(v, t);
#pragma unroll
  for (int i = 0; i < 8; ++i) xbbf[(size_t)wid*256 + i*32 + lane] = f2bf(v[i]);
  if (lane == 0) xbn[wid] = nb;
  // h1 copy + norm
#pragma unroll
  for (int i = 0; i < 8; ++i) v[i] = h1[(size_t)wid*256 + i*32 + lane];
  float hn = rnorm<8>(v);
#pragma unroll
  for (int i = 0; i < 8; ++i) h1bf[(size_t)wid*256 + i*32 + lane] = f2bf(v[i]);
  if (lane == 0) h1n[wid] = hn;
  // c1 copy + norm
#pragma unroll
  for (int i = 0; i < 8; ++i) v[i] = c1[(size_t)wid*256 + i*32 + lane];
  float cn = rnorm<8>(v);
#pragma unroll
  for (int i = 0; i < 8; ++i) c1bf[(size_t)wid*256 + i*32 + lane] = f2bf(v[i]);
  if (lane == 0) c1n[wid] = cn;
}

// ---------------- kernel: generic WMMA bf16 GEMM: C[M,NT*16] = A[M,256] * B[NT*16,256]^T ----
__global__ __launch_bounds__(256)
void gemm_bf16_k(const unsigned short* __restrict__ A, const unsigned short* __restrict__ B,
                 float* __restrict__ C, int Mtiles, int NT){
  int wave = blockIdx.x * 8 + (threadIdx.x >> 5);
  if (wave >= Mtiles * NT) return;
  int mt = wave / NT, nt = wave - mt * NT;
  int lane = threadIdx.x & 31, r = lane & 15, hi = lane >> 4;
  const unsigned short* Ar = A + (size_t)(mt*16 + r) * 256;
  const unsigned short* Br = B + (size_t)(nt*16 + r) * 256;
  v8f acc;
#pragma unroll
  for (int i = 0; i < 8; ++i) acc[i] = 0.f;
#pragma unroll
  for (int ks = 0; ks < 8; ++ks){
    int kb = ks * 32;
    acc = wmma_bf16(load_a_frag(Ar, kb, hi), load_b_frag(Br, kb, hi), acc);
  }
  int ncols = NT * 16;
  int col   = nt * 16 + r;
#pragma unroll
  for (int i = 0; i < 8; ++i)
    C[(size_t)(mt*16 + i + 8*hi) * ncols + col] = acc[i];
}

// ---------------- kernel: q = logmap0(mobius_add(mobius_matvec(W_q,x_ball), b_q)) ----------
__global__ void make_q_k(const float* __restrict__ mq, const float* __restrict__ bq,
                         const float* __restrict__ xbn, float* __restrict__ qout, int N){
  int wid  = blockIdx.x * 8 + (threadIdx.x >> 5);
  int lane = threadIdx.x & 31;
  if (wid >= N) return;
  float m[8];
#pragma unroll
  for (int i = 0; i < 8; ++i) m[i] = mq[(size_t)wid*256 + i*32 + lane];
  float mn = rnorm<8>(m);
  float nm = mv_scale<8>(m, mn, xbn[wid]);
  float yb[8];
#pragma unroll
  for (int i = 0; i < 8; ++i) yb[i] = bq[i*32 + lane];
  float y2 = rdot<8>(yb, yb);
  float xy = rdot<8>(m, yb);
  float o[8];
  mob_add<8>(m, yb, nm*nm, y2, xy, o);
  float on = rnorm<8>(o);
  on = projr<8>(o, on);
  float lq = artanh_(on) / on;
#pragma unroll
  for (int i = 0; i < 8; ++i) qout[(size_t)wid*256 + i*32 + lane] = lq * o[i];
}

// ---------------- block reduce over 256 threads ----------------
__device__ __forceinline__ float block_sum256(float v, float* sRed, float* sScal){
  int w = threadIdx.x >> 5, lane = threadIdx.x & 31;
  float s = wsum(v);
  __syncthreads();
  if (lane == 0) sRed[w] = s;
  __syncthreads();
  if (threadIdx.x == 0){
    float t = 0.f;
    for (int i = 0; i < 8; ++i) t += sRed[i];
    sScal[0] = t;
  }
  __syncthreads();
  return sScal[0];
}

// ---------------- MEGA kernel: one workgroup per node ----------------
// Phase A: 3 fused WMMA GEMMs on the 16 gathered mailbox rows -> LDS
// Phase B: per-edge mobius math -> logits + fc rows + lam_c
// Phase C: softmax over K=16
// Phase D: per-edge h path -> h_hawkes rows + lam_h
// Phase E: weighted midpoints -> h_tild (bf16 + norm), c_red (f32)
__global__ __launch_bounds__(256)
void edge_fused_k(const float* __restrict__ h1, const float* __restrict__ c1,
                  const float* __restrict__ del_t, const int* __restrict__ edges,
                  const unsigned short* __restrict__ h1bf, const unsigned short* __restrict__ c1bf,
                  const unsigned short* __restrict__ wk, const unsigned short* __restrict__ uf,
                  const unsigned short* __restrict__ wc,
                  const float* __restrict__ h1n, const float* __restrict__ c1n,
                  const float* __restrict__ q,
                  const float* __restrict__ bk, const float* __restrict__ bc,
                  const float* __restrict__ bhaw, const float* __restrict__ ahaw,
                  const float* __restrict__ haw1p, const float* __restrict__ haw2p,
                  unsigned short* __restrict__ htild_bf, float* __restrict__ htnv,
                  float* __restrict__ credw){
  __shared__ float sA[16*256];   // Wk@Hm, later h_hawkes
  __shared__ float sB[16*256];   // Uf@Hm, later fc
  __shared__ float sC[16*256];   // Wc@Cm
  __shared__ float sLog[16], sScore[16], sLamH[16], sLamC[16];
  __shared__ float sRed[8], sScal[1];

  const int n    = blockIdx.x;
  const int w    = threadIdx.x >> 5;
  const int lane = threadIdx.x & 31;
  const int r    = lane & 15;
  const int hi   = lane >> 4;

  // ---------- Phase A: GEMMs into LDS ----------
  const int srcr = edges[(size_t)n*16 + r];           // per-lane gathered A row
  const unsigned short* aH = h1bf + (size_t)srcr * 256;
  const unsigned short* aC = c1bf + (size_t)srcr * 256;
#pragma unroll
  for (int gm = 0; gm < 3; ++gm){
    const unsigned short* Ab = (gm == 2) ? aC : aH;
    const unsigned short* Wb = (gm == 0) ? wk : ((gm == 1) ? uf : wc);
    const unsigned short* b0p = Wb + (size_t)(32*w + r) * 256;
    const unsigned short* b1p = Wb + (size_t)(32*w + 16 + r) * 256;
    v8f acc0, acc1;
#pragma unroll
    for (int i = 0; i < 8; ++i){ acc0[i] = 0.f; acc1[i] = 0.f; }
#pragma unroll
    for (int ks = 0; ks < 8; ++ks){
      int kb = ks * 32;
      v16bf a = load_a_frag(Ab, kb, hi);
      acc0 = wmma_bf16(a, load_b_frag(b0p, kb, hi), acc0);
      acc1 = wmma_bf16(a, load_b_frag(b1p, kb, hi), acc1);
    }
    float* dst = (gm == 0) ? sA : ((gm == 1) ? sB : sC);
    int col = 32*w + r;
#pragma unroll
    for (int i = 0; i < 8; ++i){
      dst[(i + 8*hi)*256 + col]      = acc0[i];
      dst[(i + 8*hi)*256 + col + 16] = acc1[i];
    }
  }
  __syncthreads();

  // ---------- Phase B: per-edge logits + c path ----------
  const float Bh = bhaw[0], Ah = ahaw[0], H1s = haw1p[0], H2s = haw2p[0];
  float qv[8], ybk[8], ybc[8];
#pragma unroll
  for (int i = 0; i < 8; ++i){
    qv[i]  = q[(size_t)n*256 + i*32 + lane];
    ybk[i] = bk[i*32 + lane];
    ybc[i] = bc[i*32 + lane];
  }
  float y2k = rdot<8>(ybk, ybk);
  float y2c = rdot<8>(ybc, ybc);

  for (int e = w; e < 16; e += 8){
    int   src = edges[(size_t)n*16 + e];
    float dt  = del_t[src];
    float gt  = Bh * __expf(-Ah * dt);
    float hsn = h1n[src];
    float csn = c1n[src];

    // --- logit: h_k = mob_add(matvec(Wk,Hm), b_k); kvec = logmap0(h_k); dot q ---
    float mk[8];
#pragma unroll
    for (int i = 0; i < 8; ++i) mk[i] = sA[e*256 + i*32 + lane];
    float mkn = rnorm<8>(mk);
    float nk  = mv_scale<8>(mk, mkn, hsn);
    float xyk = rdot<8>(mk, ybk);
    float hk[8];
    mob_add<8>(mk, ybk, nk*nk, y2k, xyk, hk);
    float hn = rnorm<8>(hk);
    hn = projr<8>(hk, hn);
    float lk = artanh_(hn) / hn;
    float dq = rdot<8>(hk, qv) * lk;
    if (lane == 0) sLog[e] = dq * (1.f/16.f) + gt;

    // --- c path ---
    float mc[8];
#pragma unroll
    for (int i = 0; i < 8; ++i) mc[i] = sC[e*256 + i*32 + lane];
    float mcn = rnorm<8>(mc);
    float ncm = mv_scale<8>(mc, mcn, csn);
    float xyc = rdot<8>(mc, ybc);
    float t1v[8];
    mob_add<8>(mc, ybc, ncm*ncm, y2c, xyc, t1v);
    float t1n = rnorm<8>(t1v);
    t1n = projr<8>(t1v, t1n);
    float lt = artanh_(t1n) / t1n;
    float vv[8];
#pragma unroll
    for (int i = 0; i < 8; ++i) vv[i] = tanhf(lt * t1v[i]);
    float vn  = rnorm<8>(vv);
    float tvn = tanhf(vn);
    float se  = tvn / vn;
    float csk[8];
#pragma unroll
    for (int i = 0; i < 8; ++i) csk[i] = vv[i] * se;
    float cskn = projr<8>(csk, tvn);
    // c_sk_hat = pointwise_mul(g_t, c_sk)
    float wxn1 = fmaxf(fabsf(gt) * cskn, NFLOOR);
    float th1  = tanhf(wxn1 / fmaxf(cskn, NFLOOR) * artanh_(cskn));
    float sh1  = th1 / wxn1;
    float ckh[8];
#pragma unroll
    for (int i = 0; i < 8; ++i) ckh[i] = gt * csk[i] * sh1;
    float ckhn = projr<8>(ckh, fabsf(th1));
    // c_k_tilde = mob_add(mob_add(-c_sk, Cm), c_sk_hat)
    float cm[8], nx[8];
#pragma unroll
    for (int i = 0; i < 8; ++i){
      cm[i] = c1[(size_t)src*256 + i*32 + lane];
      nx[i] = -csk[i];
    }
    float xym = rdot<8>(nx, cm);
    float t2v[8];
    mob_add<8>(nx, cm, cskn*cskn, csn*csn, xym, t2v);
    float t2n = rnorm<8>(t2v);
    t2n = projr<8>(t2v, t2n);
    float xy3 = rdot<8>(t2v, ckh);
    float ckt[8];
    mob_add<8>(t2v, ckh, t2n*t2n, ckhn*ckhn, xy3, ckt);
    float cktn = rnorm<8>(ckt);
    cktn = projr<8>(ckt, cktn);
    // f = sigmoid(logmap0(matvec(Uf, Hm)))
    float mf[8];
#pragma unroll
    for (int i = 0; i < 8; ++i) mf[i] = sB[e*256 + i*32 + lane];
    float mfn = rnorm<8>(mf);
    float nmf = mv_scale<8>(mf, mfn, hsn);
    float lf  = artanh_(nmf) / nmf;
    float fv[8];
#pragma unroll
    for (int i = 0; i < 8; ++i) fv[i] = sigm(lf * mf[i]);
    // fc = pointwise_mul(f, c_k_tilde); lam_c
    float wx[8];
#pragma unroll
    for (int i = 0; i < 8; ++i) wx[i] = fv[i] * ckt[i];
    float wxn2 = rnorm<8>(wx);
    float fcn  = pw_scale<8>(wx, wxn2, cktn);
    float lamc = 2.f / fmaxf(1.f - fcn*fcn, EPSF);
#pragma unroll
    for (int i = 0; i < 8; ++i) sB[e*256 + i*32 + lane] = wx[i];
    if (lane == 0) sLamC[e] = lamc;
  }
  __syncthreads();

  // ---------- Phase C: softmax over 16 logits ----------
  if (threadIdx.x < 16){
    float m = -3.4e38f;
    for (int k2 = 0; k2 < 16; ++k2) m = fmaxf(m, sLog[k2]);
    float ss = 0.f;
    for (int k2 = 0; k2 < 16; ++k2) ss += __expf(sLog[k2] - m);
    sScore[threadIdx.x] = __expf(sLog[threadIdx.x] - m) / ss;
  }
  __syncthreads();

  // ---------- Phase D: per-edge h path ----------
  for (int e = w; e < 16; e += 8){
    int   src   = edges[(size_t)n*16 + e];
    float dt    = del_t[src];
    float decay = __expf(-H2s * dt * (1.f/60.f));
    float score = sScore[e];
    float hsn   = h1n[src];
    float hmv[8];
#pragma unroll
    for (int i = 0; i < 8; ++i) hmv[i] = h1[(size_t)src*256 + i*32 + lane];
    // h_scaled = pointwise_mul(score, Hm)
    float wxn = fmaxf(fabsf(score) * hsn, NFLOOR);
    float t1  = tanhf(wxn / fmaxf(hsn, NFLOOR) * artanh_(hsn));
    float s1  = t1 / wxn;
    float hs[8];
#pragma unroll
    for (int i = 0; i < 8; ++i) hs[i] = score * hmv[i] * s1;
    float hsn2 = projr<8>(hs, fabsf(t1));
    // h_max = pointwise_mul(decay, relu(h_scaled))
    float rv[8];
#pragma unroll
    for (int i = 0; i < 8; ++i) rv[i] = fmaxf(hs[i], 0.f);
    float rn   = rnorm<8>(rv);
    float wxn2 = fmaxf(decay * rn, NFLOOR);
    float t2   = tanhf(wxn2 / rn * artanh_(rn));
    float s2   = t2 / wxn2;
    float hx[8];
#pragma unroll
    for (int i = 0; i < 8; ++i) hx[i] = decay * rv[i] * s2;
    float hxn = projr<8>(hx, fabsf(t2));
    // h_hawkes = mob_add(h_scaled, haw_1 * h_max)
    float yy[8];
#pragma unroll
    for (int i = 0; i < 8; ++i) yy[i] = H1s * hx[i];
    float xy = rdot<8>(hs, yy);
    float hh[8];
    mob_add<8>(hs, yy, hsn2*hsn2, H1s*H1s*hxn*hxn, xy, hh);
    float hhn = rnorm<8>(hh);
    hhn = projr<8>(hh, hhn);
    float lam = 2.f / fmaxf(1.f - hhn*hhn, EPSF);
#pragma unroll
    for (int i = 0; i < 8; ++i) sA[e*256 + i*32 + lane] = hh[i];
    if (lane == 0) sLamH[e] = lam;
  }
  __syncthreads();

  // ---------- Phase E: weighted midpoints ----------
  int t = threadIdx.x;
  float numh = 0.f, numc = 0.f, denh = 0.f, denc = 0.f;
#pragma unroll
  for (int k2 = 0; k2 < 16; ++k2){
    float lh = sLamH[k2], lc = sLamC[k2];
    numh += lh * sA[k2*256 + t];
    numc += lc * sB[k2*256 + t];
    denh += lh - 1.f;
    denc += lc - 1.f;
  }
  denh = (fabsf(denh) < 1e-10f) ? 1e-10f : denh;
  denc = (fabsf(denc) < 1e-10f) ? 1e-10f : denc;
  float vh = numh / denh, vc = numc / denc;

  float nh2 = block_sum256(vh*vh, sRed, sScal);
  float nh  = sqrtf(fmaxf(nh2, EPSF));
  float th  = tanhf(0.5f * artanh_(nh));
  float hv  = vh * (th / nh);
  float hnn = th;
  if (hnn > MAXN){ hv *= MAXN / hnn; hnn = MAXN; }
  htild_bf[(size_t)n*256 + t] = f2bf(hv);
  if (t == 0) htnv[n] = hnn;

  float nc2 = block_sum256(vc*vc, sRed, sScal);
  float ncn = sqrtf(fmaxf(nc2, EPSF));
  float tc  = tanhf(0.5f * artanh_(ncn));
  float cv  = vc * (tc / ncn);
  if (tc > MAXN) cv *= MAXN / tc;
  credw[(size_t)n*256 + t] = cv;
}

// ---------------- kernel: final iou / gate / output ----------------
__global__ void finalize_k(const float* __restrict__ iou0r, const float* __restrict__ miour,
                           const float* __restrict__ biou,
                           const float* __restrict__ xbn, const float* __restrict__ htnv,
                           const float* __restrict__ credw,
                           float* __restrict__ out_h, float* __restrict__ out_c, int N){
  int wid  = blockIdx.x * 8 + (threadIdx.x >> 5);
  int lane = threadIdx.x & 31;
  if (wid >= N) return;

  float a[24], b[24];
#pragma unroll
  for (int j = 0; j < 24; ++j) a[j] = iou0r[(size_t)wid*768 + j*32 + lane];
  float an = rnorm<24>(a);
  float na = mv_scale<24>(a, an, xbn[wid]);               // iou0 = matvec(W_iou, x_ball)
#pragma unroll
  for (int j = 0; j < 24; ++j) b[j] = miour[(size_t)wid*768 + j*32 + lane];
  float bn = rnorm<24>(b);
  float nb = mv_scale<24>(b, bn, htnv[wid]);              // matvec(U_iou, h_tild)
  // iou1 = mob_add(iou0, m1)
  float xy = rdot<24>(a, b);
  float o1[24];
  mob_add<24>(a, b, na*na, nb*nb, xy, o1);
  float n1 = rnorm<24>(o1);
  n1 = projr<24>(o1, n1);
  // iou = mob_add(iou1, b_iou)
  float yb[24];
#pragma unroll
  for (int j = 0; j < 24; ++j) yb[j] = biou[j*32 + lane];
  float y2  = rdot<24>(yb, yb);
  float xy2 = rdot<24>(o1, yb);
  float iou[24];
  mob_add<24>(o1, yb, n1*n1, y2, xy2, iou);
  float ni = rnorm<24>(iou);
  ni = projr<24>(iou, ni);
  // split -> per-third logmap0 + gates   (element e of third t is iou[t*8+jj], e = jj*32+lane)
  float s0 = 0.f, s1 = 0.f, s2 = 0.f;
#pragma unroll
  for (int j = 0; j < 8; ++j){ s0 += iou[j]*iou[j]; s1 += iou[8+j]*iou[8+j]; s2 += iou[16+j]*iou[16+j]; }
  float n_i = sqrtf(fmaxf(wsum(s0), EPSF));
  float n_o = sqrtf(fmaxf(wsum(s1), EPSF));
  float n_u = sqrtf(fmaxf(wsum(s2), EPSF));
  float li = artanh_(n_i)/n_i, lo = artanh_(n_o)/n_o, lu = artanh_(n_u)/n_u;
  float iv[8], ov[8], uv[8];
#pragma unroll
  for (int j = 0; j < 8; ++j){
    iv[j] = sigm(li * iou[j]);
    ov[j] = sigm(lo * iou[8+j]);
    uv[j] = tanhf(lu * iou[16+j]);
  }
  // pm = mobius_pointwise_mul(i, u2)
  float un = rnorm<8>(uv);
  float wx[8];
#pragma unroll
  for (int j = 0; j < 8; ++j) wx[j] = iv[j] * uv[j];
  float wxn = rnorm<8>(wx);
  float npm = pw_scale<8>(wx, wxn, un);
  // c_new = mob_add(pm, c_red)
  float cr[8];
#pragma unroll
  for (int j = 0; j < 8; ++j) cr[j] = credw[(size_t)wid*256 + j*32 + lane];
  float y2c = rdot<8>(cr, cr);
  float xyc = rdot<8>(wx, cr);
  float cnv[8];
  mob_add<8>(wx, cr, npm*npm, y2c, xyc, cnv);
  float cn = rnorm<8>(cnv);
  cn = projr<8>(cnv, cn);
#pragma unroll
  for (int j = 0; j < 8; ++j) out_c[(size_t)wid*256 + j*32 + lane] = cnv[j];
  // h_new = mobius_pointwise_mul(o, tanh(logmap0(c_new)))
  float lc = artanh_(cn) / cn;
  float tv[8];
#pragma unroll
  for (int j = 0; j < 8; ++j) tv[j] = tanhf(lc * cnv[j]);
  float tn = rnorm<8>(tv);
  float wh[8];
#pragma unroll
  for (int j = 0; j < 8; ++j) wh[j] = ov[j] * tv[j];
  float whn = rnorm<8>(wh);
  pw_scale<8>(wh, whn, tn);
#pragma unroll
  for (int j = 0; j < 8; ++j) out_h[(size_t)wid*256 + j*32 + lane] = wh[j];
}

// ---------------- host launcher ----------------
extern "C" void kernel_launch(void* const* d_in, const int* in_sizes, int n_in,
                              void* d_out, int out_size, void* d_ws, size_t ws_size,
                              hipStream_t stream) {
  (void)in_sizes; (void)n_in; (void)out_size; (void)ws_size;
  const int N = 16384;

  const float* x     = (const float*)d_in[0];
  const float* h1    = (const float*)d_in[1];
  const float* c1    = (const float*)d_in[2];
  const float* del_t = (const float*)d_in[3];
  const int*   edges = (const int*)  d_in[4];
  const float* W_iou = (const float*)d_in[5];
  const float* U_iou = (const float*)d_in[6];
  const float* b_iou = (const float*)d_in[7];
  const float* U_f   = (const float*)d_in[8];
  const float* W_q   = (const float*)d_in[9];
  const float* b_q   = (const float*)d_in[10];
  const float* W_k   = (const float*)d_in[11];
  const float* b_k   = (const float*)d_in[12];
  const float* W_c   = (const float*)d_in[13];
  const float* b_c   = (const float*)d_in[14];
  const float* b_haw = (const float*)d_in[15];
  const float* a_haw = (const float*)d_in[16];
  const float* haw_1 = (const float*)d_in[17];
  const float* haw_2 = (const float*)d_in[18];

  // workspace partition
  char*  ws  = (char*)d_ws;
  size_t off = 0;
  auto alloc = [&](size_t bytes) -> void* {
    void* p = ws + off;
    off += (bytes + 255) & ~(size_t)255;
    return p;
  };
  unsigned short* wq_bf   = (unsigned short*)alloc((size_t)256*256*2);
  unsigned short* wk_bf   = (unsigned short*)alloc((size_t)256*256*2);
  unsigned short* wc_bf   = (unsigned short*)alloc((size_t)256*256*2);
  unsigned short* uf_bf   = (unsigned short*)alloc((size_t)256*256*2);
  unsigned short* wiou_bf = (unsigned short*)alloc((size_t)768*256*2);
  unsigned short* uiou_bf = (unsigned short*)alloc((size_t)768*256*2);
  unsigned short* h1bf    = (unsigned short*)alloc((size_t)N*256*2);
  unsigned short* c1bf    = (unsigned short*)alloc((size_t)N*256*2);
  unsigned short* xbbf    = (unsigned short*)alloc((size_t)N*256*2);
  unsigned short* htildbf = (unsigned short*)alloc((size_t)N*256*2);
  float* xbn    = (float*)alloc((size_t)N*4);
  float* h1n    = (float*)alloc((size_t)N*4);
  float* c1n    = (float*)alloc((size_t)N*4);
  float* htn    = (float*)alloc((size_t)N*4);
  float* mq_raw = (float*)alloc((size_t)N*256*4);
  float* qbuf   = (float*)alloc((size_t)N*256*4);
  float* iou0r  = (float*)alloc((size_t)N*768*4);
  float* miour  = (float*)alloc((size_t)N*768*4);
  float* credw  = (float*)alloc((size_t)N*256*4);

  float* out_h = (float*)d_out;
  float* out_c = out_h + (size_t)N*256;

  // 1. weights -> bf16
  cvt_bf16_k<<<(65536+255)/256, 256, 0, stream>>>(W_q,   wq_bf,   65536);
  cvt_bf16_k<<<(65536+255)/256, 256, 0, stream>>>(W_k,   wk_bf,   65536);
  cvt_bf16_k<<<(65536+255)/256, 256, 0, stream>>>(W_c,   wc_bf,   65536);
  cvt_bf16_k<<<(65536+255)/256, 256, 0, stream>>>(U_f,   uf_bf,   65536);
  cvt_bf16_k<<<(196608+255)/256,256, 0, stream>>>(W_iou, wiou_bf, 196608);
  cvt_bf16_k<<<(196608+255)/256,256, 0, stream>>>(U_iou, uiou_bf, 196608);

  // 2. per-node prep
  prep_nodes_k<<<N/8, 256, 0, stream>>>(x, h1, c1, xbbf, h1bf, c1bf, xbn, h1n, c1n, N);

  // 3. dense GEMMs: mq = x_ball @ Wq^T ; iou0 = x_ball @ Wiou^T
  gemm_bf16_k<<<(1024*16)/8, 256, 0, stream>>>(xbbf, wq_bf,   mq_raw, 1024, 16);
  gemm_bf16_k<<<(1024*48)/8, 256, 0, stream>>>(xbbf, wiou_bf, iou0r,  1024, 48);

  // 4. q = logmap0(mobius_add(matvec(Wq, x_ball), b_q))
  make_q_k<<<N/8, 256, 0, stream>>>(mq_raw, b_q, xbn, qbuf, N);

  // 5. mega kernel: per-node fused mailbox GEMMs + attention + midpoints
  edge_fused_k<<<N, 256, 0, stream>>>(h1, c1, del_t, edges, h1bf, c1bf,
                                      wk_bf, uf_bf, wc_bf, h1n, c1n, qbuf,
                                      b_k, b_c, b_haw, a_haw, haw_1, haw_2,
                                      htildbf, htn, credw);

  // 6. miou = h_tild @ Uiou^T
  gemm_bf16_k<<<(1024*48)/8, 256, 0, stream>>>(htildbf, uiou_bf, miour, 1024, 48);

  // 7. finalize -> h_new, c_new
  finalize_k<<<N/8, 256, 0, stream>>>(iou0r, miour, b_iou, xbn, htn, credw,
                                      out_h, out_c, N);
}
